// MoELayer_57913339019896
// MI455X (gfx1250) — compile-verified, matrix operand order checked by reading, not run on hardware
//
#include <hip/hip_runtime.h>

// ---------------------------------------------------------------------------
// MoE layer for MI455X (gfx1250, wave32).
//   B=4 T=2048 D=1024 DF=2048 E=16 K=2 C=2048
// bf16 WMMA GEMMs (v_wmma_f32_16x16x32_bf16, f32 accum), tile staging via
// Tensor Data Mover (tensor_load_to_lds + s_wait_tensorcnt), double-buffered
// LDS, gated scatter via global_atomic_add_f32. ~470 MB workspace.
// ---------------------------------------------------------------------------

static constexpr int NTOK = 8192;       // B*T
static constexpr int DIM  = 1024;       // D
static constexpr int DFF  = 2048;       // DF
static constexpr int NEXP = 16;         // E
static constexpr int TOPK = 2;          // K
static constexpr int CAP  = 2048;       // per-expert capacity
static constexpr int ECAP = NEXP * CAP; // 32768

static constexpr size_t AUX_OFF = (size_t)NTOK * DIM;   // d_out[AUX_OFF] = aux loss
static constexpr size_t IDX_OFF = AUX_OFF + 1;          // then N*K indices (as float)

typedef __attribute__((ext_vector_type(16))) __bf16 v16bf;
typedef __attribute__((ext_vector_type(8)))  __bf16 v8bf;
typedef __attribute__((ext_vector_type(8)))  float  v8f;
typedef __attribute__((ext_vector_type(4)))  unsigned int v4u;
typedef __attribute__((ext_vector_type(8)))  int v8i;
typedef __attribute__((ext_vector_type(4)))  int v4i;

union FragAB { v16bf v; v8bf h[2]; };

// ---------------------------------------------------------------------------
// TDM: DMA one [128 rows][32 cols] bf16 tile (row stride `stride_elems` in
// global) into LDS at byte offset lds_off, padding each 64B row of data with
// 16B so LDS rows land at a 40-element stride (bank-conflict padding).
// D# packing per CDNA5 ISA (group0 + group1, 2D mode).
// ---------------------------------------------------------------------------
__device__ __forceinline__ void tdm_load_tile(
    const __bf16* gptr, unsigned lds_off, unsigned tensor_d0,
    unsigned stride_elems) {
  unsigned long long ga = (unsigned long long)(uintptr_t)gptr;
  v4u g0;
  g0[0] = 1u;                                         // count=1, user mode
  g0[1] = lds_off;                                    // lds_addr
  g0[2] = (unsigned)(ga & 0xffffffffu);               // global_addr[31:0]
  g0[3] = (unsigned)((ga >> 32) & 0x01ffffffu)        // global_addr[56:32]
        | (2u << 30);                                 // type=2 (image)
  const unsigned td0 = tensor_d0;
  const unsigned td1 = 1u << 20;                      // plenty of rows
  v8i g1;
  g1[0] = (int)((1u << 16)      /* data_size=2B */
              | (1u << 20)      /* pad_enable  */
              | (3u << 22)      /* pad_interval: 16 DWORDs */
              | (3u << 25));    /* pad_amount:   4 DWORDs */
  g1[1] = (int)((td0 & 0xffffu) << 16);               // tensor_dim0[15:0]
  g1[2] = (int)((td0 >> 16) | ((td1 & 0xffffu) << 16));
  g1[3] = (int)((td1 >> 16) | (32u << 16));           // tile_dim0 = 32
  g1[4] = (int)(128u);                                // tile_dim1 = 128
  g1[5] = (int)stride_elems;                          // tensor_dim0_stride
  g1[6] = 0;
  g1[7] = 0;
  v4i g2 = {};
  v4i g3 = {};
#if defined(__clang_major__) && __clang_major__ >= 23
  v8i g4 = {};
  __builtin_amdgcn_tensor_load_to_lds(g0, g1, g2, g3, g4, 0);
#else
  __builtin_amdgcn_tensor_load_to_lds(g0, g1, g2, g3, 0);
#endif
}

__device__ __forceinline__ unsigned lds_offset_of(const void* p) {
  // generic pointers to LDS carry the LDS byte offset in the low 32 bits
  return (unsigned)(uintptr_t)p;
}

// ---------------------------------------------------------------------------
// f32 -> bf16 streaming convert (4 elems / thread)
// ---------------------------------------------------------------------------
__global__ __launch_bounds__(256) void f32_to_bf16_kernel(
    const float* __restrict__ src, __bf16* __restrict__ dst, long n) {
  long i = ((long)blockIdx.x * blockDim.x + threadIdx.x) * 4;
  if (i + 3 < n) {
    float4 v = *(const float4*)&src[i];
    dst[i + 0] = (__bf16)v.x;
    dst[i + 1] = (__bf16)v.y;
    dst[i + 2] = (__bf16)v.z;
    dst[i + 3] = (__bf16)v.w;
  }
}

// ---------------------------------------------------------------------------
// Transpose + convert: dst[c][r] = (bf16)src[r][c]   (dst is [cols][rows])
// grid = (cols/32, rows/32, batch), block = 256 (32x8)
// ---------------------------------------------------------------------------
__global__ __launch_bounds__(256) void tr_conv_kernel(
    const float* __restrict__ src, __bf16* __restrict__ dst, int rows, int cols) {
  __shared__ float tile[32][33];
  const long b = blockIdx.z;
  src += b * (long)rows * cols;
  dst += b * (long)rows * cols;
  const int tx = threadIdx.x & 31, ty = threadIdx.x >> 5;
  const int r0 = blockIdx.y * 32, c0 = blockIdx.x * 32;
#pragma unroll
  for (int j = 0; j < 4; ++j) {
    int r = r0 + ty + j * 8;
    tile[ty + j * 8][tx] = src[(long)r * cols + c0 + tx];
  }
  __syncthreads();
#pragma unroll
  for (int j = 0; j < 4; ++j) {
    int c = c0 + ty + j * 8;
    dst[(long)c * rows + r0 + tx] = (__bf16)tile[tx][ty + j * 8];
  }
}

// ---------------------------------------------------------------------------
// Init: invalidate slot table, clear gates/counters
// ---------------------------------------------------------------------------
__global__ __launch_bounds__(256) void init_kernel(
    int* __restrict__ tok_slot, float* __restrict__ gate_slot,
    int* __restrict__ counts) {
  int gid = blockIdx.x * blockDim.x + threadIdx.x;
  if (gid < ECAP) { tok_slot[gid] = NTOK; gate_slot[gid] = 0.0f; }
  if (gid < NEXP) counts[gid] = 0;
}

// ---------------------------------------------------------------------------
// Router: one wave per token. Lanes 0..15 each own one expert.
// ---------------------------------------------------------------------------
__global__ __launch_bounds__(256) void router_kernel(
    const float* __restrict__ x, const float* __restrict__ rw,
    const float* __restrict__ rb, int* __restrict__ eid,
    float* __restrict__ gprob, float* __restrict__ out) {
  const int wave = threadIdx.x >> 5, lane = threadIdx.x & 31;
  const int n = blockIdx.x * 8 + wave;
  float acc = 0.0f;
  if (lane < NEXP) {
    const float* xr = x + (size_t)n * DIM;
    for (int d = 0; d < DIM; ++d)
      acc = fmaf(xr[d], rw[d * NEXP + lane], acc);
    acc += rb[lane];
  }
  float lg[NEXP];
#pragma unroll
  for (int e = 0; e < NEXP; ++e) lg[e] = __shfl(acc, e, 32);

  float mx = lg[0];
#pragma unroll
  for (int e = 1; e < NEXP; ++e) mx = fmaxf(mx, lg[e]);
  float p[NEXP], sum = 0.0f;
#pragma unroll
  for (int e = 0; e < NEXP; ++e) { p[e] = __expf(lg[e] - mx); sum += p[e]; }
  const float inv = 1.0f / sum;

  int i1 = 0; float b1 = p[0];
#pragma unroll
  for (int e = 1; e < NEXP; ++e) if (p[e] > b1) { b1 = p[e]; i1 = e; }
  int i2 = (i1 == 0) ? 1 : 0; float b2 = p[i2];
#pragma unroll
  for (int e = 0; e < NEXP; ++e)
    if (e != i1 && p[e] > b2) { b2 = p[e]; i2 = e; }

  const float p1 = b1 * inv, p2 = b2 * inv;
  const float den = p1 + p2 + 1e-9f;
  const float g1 = p1 / den, g2 = p2 / den;

  if (lane == 0) {
    eid[n * TOPK + 0] = i1; eid[n * TOPK + 1] = i2;
    gprob[n * TOPK + 0] = g1; gprob[n * TOPK + 1] = g2;
    out[IDX_OFF + n * TOPK + 0] = (float)i1;
    out[IDX_OFF + n * TOPK + 1] = (float)i2;
    if (n == 0) out[AUX_OFF] = 0.0f;
  }
}

// ---------------------------------------------------------------------------
// Capacity assignment
// ---------------------------------------------------------------------------
__global__ __launch_bounds__(256) void assign_kernel(
    const int* __restrict__ eid, const float* __restrict__ gprob,
    int* __restrict__ counts, int* __restrict__ tok_slot,
    float* __restrict__ gate_slot) {
  int gid = blockIdx.x * blockDim.x + threadIdx.x;
  if (gid >= NTOK * TOPK) return;
  int e = eid[gid];
  int slot = atomicAdd(&counts[e], 1);
  if (slot < CAP) {
    tok_slot[e * CAP + slot] = gid >> 1;
    gate_slot[e * CAP + slot] = gprob[gid];
  }
}

// ---------------------------------------------------------------------------
// Gather tokens (bf16) into slot-ordered expert input; zero invalid slots.
// ---------------------------------------------------------------------------
__global__ __launch_bounds__(256) void gather_kernel(
    const __bf16* __restrict__ xb, const int* __restrict__ tok_slot,
    __bf16* __restrict__ xe) {
  const int row = blockIdx.x * 2 + (threadIdx.x >> 7);
  const int t = threadIdx.x & 127;
  const int tok = tok_slot[row];
  v8bf v = {};
  if (tok < NTOK) v = *(const v8bf*)&xb[(size_t)tok * DIM + t * 8];
  *(v8bf*)&xe[(size_t)row * DIM + t * 8] = v;
}

// ---------------------------------------------------------------------------
// Fused gate/up GEMM + SiLU:  H = silu(X @ Wg^T) * (X @ Wu^T)
//   X: [M][DIM] bf16, Wg/Wu: [DFF][DIM] bf16 (N-major!), H: [M][DFF] bf16
// All three tiles are identical [128][32] row-tiles with stride DIM, staged
// by TDM into double-buffered LDS (row stride 40 elems via TDM padding).
// Block tile 128x128, 8 waves (2x4), wave tile 64x32.
// Fragment layout (16-bit WMMA): lanes 0-15 K={0..7,16..23},
// lanes 16-31 K={8..15,24..31}; each fragment = two b128 LDS loads.
// ---------------------------------------------------------------------------
__global__ __launch_bounds__(256) void gemm_gateup_kernel(
    const __bf16* __restrict__ Xb, const __bf16* __restrict__ Wg,
    const __bf16* __restrict__ Wu, __bf16* __restrict__ H,
    long xBatch, long wBatch, long hBatch) {
  constexpr int LDT = 40;
  __shared__ __align__(16) __bf16 Xs[2][128 * LDT];
  __shared__ __align__(16) __bf16 Gs[2][128 * LDT];
  __shared__ __align__(16) __bf16 Us[2][128 * LDT];

  const long z = blockIdx.z;
  Xb += z * xBatch; Wg += z * wBatch; Wu += z * wBatch; H += z * hBatch;

  const int bm0 = blockIdx.y * 128, bn0 = blockIdx.x * 128;
  const int tid = threadIdx.x, lane = tid & 31, wave = tid >> 5;
  const int half = lane >> 4, l16 = lane & 15;
  const int wm = (wave >> 2) * 64, wn = (wave & 3) * 32;

  const __bf16* xTile = Xb + (size_t)bm0 * DIM;
  const __bf16* gTile = Wg + (size_t)bn0 * DIM;
  const __bf16* uTile = Wu + (size_t)bn0 * DIM;

  v8f cg[4][2] = {};
  v8f cu[4][2] = {};

  if (wave == 0) {
    tdm_load_tile(xTile, lds_offset_of(&Xs[0][0]), DIM, DIM);
    tdm_load_tile(gTile, lds_offset_of(&Gs[0][0]), DIM, DIM);
    tdm_load_tile(uTile, lds_offset_of(&Us[0][0]), DIM, DIM);
  }

  int cur = 0;
  for (int kt = 0; kt < DIM / 32; ++kt) {
    const int kk = kt * 32;
    const bool more = (kt + 1) < (DIM / 32);
    if (wave == 0) {
      if (more) {
        const int nk = kk + 32;
        tdm_load_tile(xTile + nk, lds_offset_of(&Xs[cur ^ 1][0]), DIM, DIM);
        tdm_load_tile(gTile + nk, lds_offset_of(&Gs[cur ^ 1][0]), DIM, DIM);
        tdm_load_tile(uTile + nk, lds_offset_of(&Us[cur ^ 1][0]), DIM, DIM);
        __builtin_amdgcn_s_wait_tensorcnt(3);  // current buffer's 3 DMAs done
      } else {
        __builtin_amdgcn_s_wait_tensorcnt(0);
      }
    }
    __syncthreads();

    FragAB a[4], bg[2], bu[2];
#pragma unroll
    for (int mi = 0; mi < 4; ++mi) {
      int row = wm + mi * 16 + l16;
      a[mi].h[0] = *(const v8bf*)&Xs[cur][row * LDT + 8 * half];
      a[mi].h[1] = *(const v8bf*)&Xs[cur][row * LDT + 16 + 8 * half];
    }
#pragma unroll
    for (int ni = 0; ni < 2; ++ni) {
      int col = wn + ni * 16 + l16;
      bg[ni].h[0] = *(const v8bf*)&Gs[cur][col * LDT + 8 * half];
      bg[ni].h[1] = *(const v8bf*)&Gs[cur][col * LDT + 16 + 8 * half];
      bu[ni].h[0] = *(const v8bf*)&Us[cur][col * LDT + 8 * half];
      bu[ni].h[1] = *(const v8bf*)&Us[cur][col * LDT + 16 + 8 * half];
    }
#pragma unroll
    for (int mi = 0; mi < 4; ++mi)
#pragma unroll
      for (int ni = 0; ni < 2; ++ni) {
        cg[mi][ni] = __builtin_amdgcn_wmma_f32_16x16x32_bf16(
            false, a[mi].v, false, bg[ni].v, (short)0, cg[mi][ni], false, false);
        cu[mi][ni] = __builtin_amdgcn_wmma_f32_16x16x32_bf16(
            false, a[mi].v, false, bu[ni].v, (short)0, cu[mi][ni], false, false);
      }
    __syncthreads();  // everyone done reading buf before TDM overwrites it
    cur ^= 1;
  }

#pragma unroll
  for (int mi = 0; mi < 4; ++mi)
#pragma unroll
    for (int ni = 0; ni < 2; ++ni)
#pragma unroll
      for (int r = 0; r < 8; ++r) {
        int m = bm0 + wm + mi * 16 + half * 8 + r;
        int col = bn0 + wn + ni * 16 + l16;
        float g = cg[mi][ni][r];
        float u = cu[mi][ni][r];
        float hval = (g / (1.0f + __expf(-g))) * u;
        H[(size_t)m * DFF + col] = (__bf16)hval;
      }
}

// ---------------------------------------------------------------------------
// Down GEMM:  Out = H @ Wd^T
//   H: [M][DFF] bf16, Wd: [DIM][DFF] bf16 (N-major), Out: f32
// scatter==0: plain store; scatter==1: gated global_atomic_add_f32 scatter.
// ---------------------------------------------------------------------------
__global__ __launch_bounds__(256) void gemm_down_kernel(
    const __bf16* __restrict__ Hsrc, const __bf16* __restrict__ Wd,
    float* __restrict__ Out, const int* __restrict__ tok_slot,
    const float* __restrict__ gate_slot, int scatter,
    long hBatch, long wBatch) {
  constexpr int LDT = 40;
  __shared__ __align__(16) __bf16 Hs[2][128 * LDT];
  __shared__ __align__(16) __bf16 Ws[2][128 * LDT];

  const long z = blockIdx.z;
  Hsrc += z * hBatch; Wd += z * wBatch;

  const int bm0 = blockIdx.y * 128, bn0 = blockIdx.x * 128;
  const int tid = threadIdx.x, lane = tid & 31, wave = tid >> 5;
  const int half = lane >> 4, l16 = lane & 15;
  const int wm = (wave >> 2) * 64, wn = (wave & 3) * 32;

  const __bf16* hTile = Hsrc + (size_t)bm0 * DFF;
  const __bf16* wTile = Wd + (size_t)bn0 * DFF;

  v8f cd[4][2] = {};

  if (wave == 0) {
    tdm_load_tile(hTile, lds_offset_of(&Hs[0][0]), DFF, DFF);
    tdm_load_tile(wTile, lds_offset_of(&Ws[0][0]), DFF, DFF);
  }

  int cur = 0;
  for (int kt = 0; kt < DFF / 32; ++kt) {
    const int kk = kt * 32;
    const bool more = (kt + 1) < (DFF / 32);
    if (wave == 0) {
      if (more) {
        const int nk = kk + 32;
        tdm_load_tile(hTile + nk, lds_offset_of(&Hs[cur ^ 1][0]), DFF, DFF);
        tdm_load_tile(wTile + nk, lds_offset_of(&Ws[cur ^ 1][0]), DFF, DFF);
        __builtin_amdgcn_s_wait_tensorcnt(2);
      } else {
        __builtin_amdgcn_s_wait_tensorcnt(0);
      }
    }
    __syncthreads();

    FragAB a[4], b[2];
#pragma unroll
    for (int mi = 0; mi < 4; ++mi) {
      int row = wm + mi * 16 + l16;
      a[mi].h[0] = *(const v8bf*)&Hs[cur][row * LDT + 8 * half];
      a[mi].h[1] = *(const v8bf*)&Hs[cur][row * LDT + 16 + 8 * half];
    }
#pragma unroll
    for (int ni = 0; ni < 2; ++ni) {
      int col = wn + ni * 16 + l16;
      b[ni].h[0] = *(const v8bf*)&Ws[cur][col * LDT + 8 * half];
      b[ni].h[1] = *(const v8bf*)&Ws[cur][col * LDT + 16 + 8 * half];
    }
#pragma unroll
    for (int mi = 0; mi < 4; ++mi)
#pragma unroll
      for (int ni = 0; ni < 2; ++ni)
        cd[mi][ni] = __builtin_amdgcn_wmma_f32_16x16x32_bf16(
            false, a[mi].v, false, b[ni].v, (short)0, cd[mi][ni], false, false);
    __syncthreads();
    cur ^= 1;
  }

#pragma unroll
  for (int mi = 0; mi < 4; ++mi)
#pragma unroll
    for (int ni = 0; ni < 2; ++ni)
#pragma unroll
      for (int r = 0; r < 8; ++r) {
        int mrow = bm0 + wm + mi * 16 + half * 8 + r;
        int col = bn0 + wn + ni * 16 + l16;
        float v = cd[mi][ni][r];
        if (!scatter) {
          Out[(size_t)mrow * DIM + col] = v;
        } else {
          int slot = (int)z * CAP + mrow;
          int tok = tok_slot[slot];
          if (tok < NTOK)
            atomicAdd(&Out[(size_t)tok * DIM + col], v * gate_slot[slot]);
        }
      }
}

// ---------------------------------------------------------------------------
// Host-side orchestration
// ---------------------------------------------------------------------------
extern "C" void kernel_launch(void* const* d_in, const int* in_sizes, int n_in,
                              void* d_out, int out_size, void* d_ws, size_t ws_size,
                              hipStream_t stream) {
  const float* x       = (const float*)d_in[0];
  const float* rw      = (const float*)d_in[1];
  const float* rb      = (const float*)d_in[2];
  const float* ws_gate = (const float*)d_in[3];
  const float* ws_up   = (const float*)d_in[4];
  const float* ws_down = (const float*)d_in[5];
  const float* w_gate  = (const float*)d_in[6];
  const float* w_up    = (const float*)d_in[7];
  const float* w_down  = (const float*)d_in[8];
  float* out = (float*)d_out;

  // workspace layout (~466 MB)
  size_t o = 0;
  auto take = [&](size_t bytes) -> void* {
    void* p = (char*)d_ws + o;
    o = (o + bytes + 255) & ~(size_t)255;
    return p;
  };
  __bf16* xb  = (__bf16*)take((size_t)NTOK * DIM * 2);        // x bf16 [N][D]
  __bf16* wgb = (__bf16*)take((size_t)DFF * DIM * 2);         // shared gate [DF][D]
  __bf16* wub = (__bf16*)take((size_t)DFF * DIM * 2);         // shared up   [DF][D]
  __bf16* wdb = (__bf16*)take((size_t)DIM * DFF * 2);         // shared down [D][DF]
  __bf16* wge = (__bf16*)take((size_t)NEXP * DFF * DIM * 2);  // experts gate [E][DF][D]
  __bf16* wue = (__bf16*)take((size_t)NEXP * DFF * DIM * 2);  // experts up   [E][DF][D]
  __bf16* wde = (__bf16*)take((size_t)NEXP * DIM * DFF * 2);  // experts down [E][D][DF]
  __bf16* hs  = (__bf16*)take((size_t)NTOK * DFF * 2);        // shared hidden
  __bf16* he  = (__bf16*)take((size_t)ECAP * DFF * 2);        // expert hidden
  __bf16* xe  = (__bf16*)take((size_t)ECAP * DIM * 2);        // gathered expert input
  int*    tok_slot  = (int*)take((size_t)ECAP * 4);
  float*  gate_slot = (float*)take((size_t)ECAP * 4);
  int*    eid       = (int*)take((size_t)NTOK * TOPK * 4);
  float*  gprob     = (float*)take((size_t)NTOK * TOPK * 4);
  int*    counts    = (int*)take((size_t)NEXP * 4);
  (void)in_sizes; (void)n_in; (void)out_size; (void)ws_size;

  // 1) weight/activation conversion
  //    shared weights need transpose to N-major; expert weights are already
  //    N-major for our GEMM formulation -> plain convert.
  f32_to_bf16_kernel<<<(NTOK * DIM / 4 + 255) / 256, 256, 0, stream>>>(x, xb, (long)NTOK * DIM);
  tr_conv_kernel<<<dim3(DFF / 32, DIM / 32, 1), 256, 0, stream>>>(ws_gate, wgb, DIM, DFF);
  tr_conv_kernel<<<dim3(DFF / 32, DIM / 32, 1), 256, 0, stream>>>(ws_up,   wub, DIM, DFF);
  tr_conv_kernel<<<dim3(DIM / 32, DFF / 32, 1), 256, 0, stream>>>(ws_down, wdb, DFF, DIM);
  f32_to_bf16_kernel<<<((long)NEXP * DFF * DIM / 4 + 255) / 256, 256, 0, stream>>>(
      w_gate, wge, (long)NEXP * DFF * DIM);
  f32_to_bf16_kernel<<<((long)NEXP * DFF * DIM / 4 + 255) / 256, 256, 0, stream>>>(
      w_up, wue, (long)NEXP * DFF * DIM);
  f32_to_bf16_kernel<<<((long)NEXP * DIM * DFF / 4 + 255) / 256, 256, 0, stream>>>(
      w_down, wde, (long)NEXP * DIM * DFF);

  // 2) routing
  init_kernel<<<ECAP / 256, 256, 0, stream>>>(tok_slot, gate_slot, counts);
  router_kernel<<<NTOK / 8, 256, 0, stream>>>(x, rw, rb, eid, gprob, out);
  assign_kernel<<<(NTOK * TOPK) / 256, 256, 0, stream>>>(eid, gprob, counts, tok_slot, gate_slot);
  gather_kernel<<<ECAP / 2, 256, 0, stream>>>(xb, tok_slot, xe);

  // 3) shared expert
  gemm_gateup_kernel<<<dim3(DFF / 128, NTOK / 128, 1), 256, 0, stream>>>(
      xb, wgb, wub, hs, 0, 0, 0);
  gemm_down_kernel<<<dim3(DIM / 128, NTOK / 128, 1), 256, 0, stream>>>(
      hs, wdb, out, tok_slot, gate_slot, 0, 0, 0);

  // 4) expert bank (batched over E) + gated atomic scatter
  gemm_gateup_kernel<<<dim3(DFF / 128, CAP / 128, NEXP), 256, 0, stream>>>(
      xe, wge, wue, he, (long)CAP * DIM, (long)DFF * DIM, (long)CAP * DFF);
  gemm_down_kernel<<<dim3(DIM / 128, CAP / 128, NEXP), 256, 0, stream>>>(
      he, wde, out, tok_slot, gate_slot, 1, (long)CAP * DFF, (long)DFF * DIM);
}
